// LinearTemporalCrossAttention_781684048569
// MI455X (gfx1250) — matrix operand order repaired and checked by reading, not compiled, not run on hardware
//
#include <hip/hip_runtime.h>

// ---------------- shapes ----------------
#define BB   4
#define TT   8192
#define DD   512
#define NN   256
#define LL   768
#define TE   2048
#define HH   8
#define DH   64
#define AST  520   // LDS bf16 A-buffer row stride (260 dwords == 4 mod 8 -> conflict-free b128)
#define ROWS 64    // tokens per workgroup in fused_main
#define MT   4     // 16-row M-tiles per workgroup
#define FQS  516   // f32 LDS row stride (516 % 64 == 4 -> conflict-free strided access)

typedef __attribute__((ext_vector_type(16))) __bf16 v16bf;
typedef __attribute__((ext_vector_type(8)))  float  v8f;

union FragU { v16bf v; uint4 q[2]; };

__device__ __forceinline__ unsigned short f2bf(float f) {
  unsigned int u = __float_as_uint(f);
  u += 0x7FFFu + ((u >> 16) & 1u);   // round-to-nearest-even
  return (unsigned short)(u >> 16);
}

// A fragment: p points at element (row, kbase + asel); elems 0..15 = p[0..15] contiguous
__device__ __forceinline__ v16bf load_fragA(const unsigned short* p) {
  FragU f;
  f.q[0] = *(const uint4*)(p);
  f.q[1] = *(const uint4*)(p + 8);
  return f.v;
}
// B fragment: p points at element (col, kbase + bsel); elems 0..15 = p[0..15] contiguous
__device__ __forceinline__ v16bf load_fragB(const unsigned short* p) {
  FragU f;
  f.q[0] = *(const uint4*)(p);
  f.q[1] = *(const uint4*)(p + 8);
  return f.v;
}

__device__ __forceinline__ v8f wmma_bf16(v16bf a, v16bf b, v8f c) {
  return __builtin_amdgcn_wmma_f32_16x16x32_bf16(false, a, false, b, (short)0, c, false, false);
}

// ---------------- prep: transpose weight (K,N) f32 -> (N,K) bf16 ----------------
__global__ __launch_bounds__(256) void k_transpose_bf16(const float* __restrict__ W,
                                                        unsigned short* __restrict__ WT,
                                                        int K, int N) {
  int idx = blockIdx.x * 256 + threadIdx.x;
  if (idx >= K * N) return;
  int n = idx / K;
  int k = idx - n * K;
  WT[idx] = f2bf(W[(size_t)k * N + n]);
}

// ---------------- eo = silu(emb) @ emb_W + emb_b  [B,1024] ----------------
__global__ __launch_bounds__(256) void k_emb(const float* __restrict__ emb,
                                             const float* __restrict__ embW,
                                             const float* __restrict__ embB,
                                             float* __restrict__ eo) {
  int idx = blockIdx.x * 256 + threadIdx.x;     // 4096 total
  int b = idx >> 10, j = idx & 1023;
  const float* e = emb + (size_t)b * TE;
  float acc = 0.f;
  for (int i = 0; i < TE; ++i) {
    float ev = e[i];
    float s  = ev / (1.0f + __expf(-ev));
    acc += s * embW[(size_t)i * 1024 + j];
  }
  eo[idx] = acc + embB[j];
}

// ---------------- xfn = LN(xf) -> bf16 [B*N, 768] ----------------
__global__ __launch_bounds__(256) void k_xfn(const float* __restrict__ xf,
                                             const float* __restrict__ tg,
                                             const float* __restrict__ tb,
                                             unsigned short* __restrict__ xfn) {
  __shared__ float s1[256];
  __shared__ float s2[256];
  const int r = blockIdx.x;            // 0..1023
  const int tid = threadIdx.x;
  const float* row = xf + (size_t)r * LL;
  float v[3], s = 0.f, ss = 0.f;
  #pragma unroll
  for (int j = 0; j < 3; ++j) { float t = row[tid + 256 * j]; v[j] = t; s += t; ss += t * t; }
  s1[tid] = s; s2[tid] = ss;
  __syncthreads();
  for (int st = 128; st > 0; st >>= 1) {
    if (tid < st) { s1[tid] += s1[tid + st]; s2[tid] += s2[tid + st]; }
    __syncthreads();
  }
  float mean = s1[0] * (1.0f / LL);
  float var  = s2[0] * (1.0f / LL) - mean * mean;
  float inv  = rsqrtf(var + 1e-5f);
  unsigned short* orow = xfn + (size_t)r * LL;
  #pragma unroll
  for (int j = 0; j < 3; ++j) {
    int c = tid + 256 * j;
    orow[c] = f2bf((v[j] - mean) * inv * tg[c] + tb[c]);
  }
}

// ---------------- k/v = xfn @ W{k,v} + b{k,v}  (WMMA, pipelined) -> f32 [B*N,512] ----------------
__global__ __launch_bounds__(256) void k_kv(const unsigned short* __restrict__ xfn,
                                            const unsigned short* __restrict__ WkT,
                                            const unsigned short* __restrict__ WvT,
                                            const float* __restrict__ bk,
                                            const float* __restrict__ bv,
                                            float* __restrict__ kbuf,
                                            float* __restrict__ vbuf) {
  const int tid  = threadIdx.x;
  const int lane = tid & 31;
  const int wav  = tid >> 5;
  const int rsel = lane & 15;
  const int hi   = lane >> 4;
  const int asel = hi << 4;
  const int bsel = hi << 4;
  const int mbase = hi << 3;
  const int r0 = blockIdx.x * 16;
  const bool isV = (blockIdx.y != 0);
  const unsigned short* Wt = isV ? WvT : WkT;
  const float* bias = isV ? bv : bk;
  float* outb = isV ? vbuf : kbuf;

  v8f acc[4] = {};
  int ncol[4];
  #pragma unroll
  for (int i = 0; i < 4; ++i) ncol[i] = (wav + 8 * i) * 16 + rsel;

  const unsigned short* arow = xfn + (size_t)(r0 + rsel) * LL + asel;
  const unsigned short* bcol[4];
  #pragma unroll
  for (int i = 0; i < 4; ++i) bcol[i] = Wt + (size_t)ncol[i] * LL + bsel;

  v16bf aP, aQ, bP[4], bQ[4];
  aP = load_fragA(arow);
  #pragma unroll
  for (int i = 0; i < 4; ++i) bP[i] = load_fragB(bcol[i]);

  for (int kc = 0; kc < LL / 32; kc += 2) {
    const int kb1 = (kc + 1) * 32;
    aQ = load_fragA(arow + kb1);
    #pragma unroll
    for (int i = 0; i < 4; ++i) bQ[i] = load_fragB(bcol[i] + kb1);
    #pragma unroll
    for (int i = 0; i < 4; ++i) acc[i] = wmma_bf16(aP, bP[i], acc[i]);
    if (kc + 2 < LL / 32) {
      const int kb2 = (kc + 2) * 32;
      aP = load_fragA(arow + kb2);
      #pragma unroll
      for (int i = 0; i < 4; ++i) bP[i] = load_fragB(bcol[i] + kb2);
    }
    #pragma unroll
    for (int i = 0; i < 4; ++i) acc[i] = wmma_bf16(aQ, bQ[i], acc[i]);
  }
  #pragma unroll
  for (int i = 0; i < 4; ++i) {
    float bb = bias[ncol[i]];
    #pragma unroll
    for (int r = 0; r < 8; ++r)
      outb[(size_t)(r0 + mbase + r) * DD + ncol[i]] = acc[i][r] + bb;
  }
}

// ---------------- softmax of k over tokens (axis=1), in place ----------------
__global__ __launch_bounds__(256) void k_softk(float* __restrict__ kbuf) {
  int idx = blockIdx.x * 256 + threadIdx.x;   // 2048 = B * 512
  int b = idx >> 9, c = idx & 511;
  float* base = kbuf + (size_t)b * NN * DD + c;
  float mx = -3.0e38f;
  for (int n = 0; n < NN; ++n) mx = fmaxf(mx, base[(size_t)n * DD]);
  float s = 0.f;
  for (int n = 0; n < NN; ++n) s += __expf(base[(size_t)n * DD] - mx);
  float inv = 1.0f / s;
  for (int n = 0; n < NN; ++n) base[(size_t)n * DD] = __expf(base[(size_t)n * DD] - mx) * inv;
}

// ---------------- attT[b][h][l][d] = sum_n v[n,l] * k[n,d]  -> bf16 ----------------
__global__ __launch_bounds__(256) void k_att(const float* __restrict__ kbuf,
                                             const float* __restrict__ vbuf,
                                             unsigned short* __restrict__ attT) {
  int idx = blockIdx.x * 256 + threadIdx.x;   // 131072 = B*H*64*64
  int d = idx & 63, l = (idx >> 6) & 63, h = (idx >> 12) & 7, b = idx >> 15;
  const float* kp = kbuf + (size_t)b * NN * DD + h * DH + d;
  const float* vp = vbuf + (size_t)b * NN * DD + h * DH + l;
  float acc = 0.f;
  for (int n = 0; n < NN; ++n) acc += vp[(size_t)n * DD] * kp[(size_t)n * DD];
  attT[idx] = f2bf(acc);
}

// ---------------- fused main: 64-token tile, pipelined WMMA GEMMs ----------------
__global__ __launch_bounds__(256) void fused_main(
    const float* __restrict__ x,
    const float* __restrict__ bq,
    const float* __restrict__ ln_g, const float* __restrict__ ln_b,
    const float* __restrict__ sn_g, const float* __restrict__ sn_b,
    const float* __restrict__ out_b,
    const unsigned short* __restrict__ WqT,
    const unsigned short* __restrict__ OutWT,
    const unsigned short* __restrict__ attT,
    const float* __restrict__ eo,
    float* __restrict__ out) {

  __shared__ __align__(16) unsigned short Abf[ROWS * AST]; // 66.5 KB bf16 staging
  __shared__ __align__(16) float Fq[ROWS * FQS];           // 132 KB f32 (x -> q -> y)

  const int tid  = threadIdx.x;
  const int lane = tid & 31;
  const int wav  = tid >> 5;
  const int rsel = lane & 15;
  const int hilo = lane >> 4;
  const int asel = hilo << 4;   // A-fragment K sub-base
  const int bsel = hilo << 4;   // B-fragment K sub-base
  const int mbase = hilo << 3;  // C/D row sub-base

  const int row0 = blockIdx.x * ROWS;   // flat token index (b*T + t), 512 blocks
  const int b    = row0 / TT;

  // ---- step 0: coalesced copy x tile -> Fq (float4) ----
  {
    const float4* xv = (const float4*)(x + (size_t)row0 * DD);
    #pragma unroll
    for (int j = 0; j < 32; ++j) {
      int idx4 = tid + 256 * j;           // 0..8191  (64 rows x 128 float4)
      int r = idx4 >> 7;
      int c4 = idx4 & 127;
      *(float4*)&Fq[r * FQS + c4 * 4] = xv[idx4];
    }
  }
  __syncthreads();

  // ---- step 1: LN(x) -> Abf (bf16); 4 lanes per row ----
  {
    const int r = tid >> 2, seg = tid & 3;
    const float* fr = Fq + r * FQS;
    float s = 0.f, ss = 0.f;
    for (int j = 0; j < 128; ++j) { float v = fr[seg + 4 * j]; s += v; ss += v * v; }
    s += __shfl_xor(s, 1, 4); ss += __shfl_xor(ss, 1, 4);
    s += __shfl_xor(s, 2, 4); ss += __shfl_xor(ss, 2, 4);
    float mean = s * (1.0f / DD);
    float var  = ss * (1.0f / DD) - mean * mean;
    float inv  = rsqrtf(var + 1e-5f);
    unsigned short* ar = Abf + r * AST;
    for (int j = 0; j < 128; ++j) {
      int c = seg + 4 * j;
      ar[c] = f2bf((fr[c] - mean) * inv * ln_g[c] + ln_b[c]);
    }
  }
  __syncthreads();

  // ---- step 2: q = LN(x) @ Wq + bq (WMMA, ping-pong pipelined) -> Fq ----
  for (int pass = 0; pass < 2; ++pass) {
    v8f acc[2][MT] = {};
    int ncol[2];
    #pragma unroll
    for (int i = 0; i < 2; ++i) ncol[i] = (wav + 8 * (pass * 2 + i)) * 16 + rsel;
    const unsigned short* arow0 = Abf + rsel * AST + asel;
    const unsigned short* bp[2];
    #pragma unroll
    for (int i = 0; i < 2; ++i) bp[i] = WqT + (size_t)ncol[i] * DD + bsel;

    v16bf aP[MT], aQ[MT], bP[2], bQ[2];
    #pragma unroll
    for (int mt = 0; mt < MT; ++mt) aP[mt] = load_fragA(arow0 + mt * 16 * AST);
    #pragma unroll
    for (int i = 0; i < 2; ++i) bP[i] = load_fragB(bp[i]);

    for (int kc = 0; kc < DD / 32; kc += 2) {
      const int kb1 = (kc + 1) * 32;
      #pragma unroll
      for (int mt = 0; mt < MT; ++mt) aQ[mt] = load_fragA(arow0 + mt * 16 * AST + kb1);
      #pragma unroll
      for (int i = 0; i < 2; ++i) bQ[i] = load_fragB(bp[i] + kb1);
      #pragma unroll
      for (int i = 0; i < 2; ++i)
        #pragma unroll
        for (int mt = 0; mt < MT; ++mt) acc[i][mt] = wmma_bf16(aP[mt], bP[i], acc[i][mt]);
      if (kc + 2 < DD / 32) {
        const int kb2 = (kc + 2) * 32;
        #pragma unroll
        for (int mt = 0; mt < MT; ++mt) aP[mt] = load_fragA(arow0 + mt * 16 * AST + kb2);
        #pragma unroll
        for (int i = 0; i < 2; ++i) bP[i] = load_fragB(bp[i] + kb2);
      }
      #pragma unroll
      for (int i = 0; i < 2; ++i)
        #pragma unroll
        for (int mt = 0; mt < MT; ++mt) acc[i][mt] = wmma_bf16(aQ[mt], bQ[i], acc[i][mt]);
    }
    #pragma unroll
    for (int i = 0; i < 2; ++i) {
      float bias = bq[ncol[i]];
      #pragma unroll
      for (int mt = 0; mt < MT; ++mt)
        #pragma unroll
        for (int r = 0; r < 8; ++r)
          Fq[(mt * 16 + mbase + r) * FQS + ncol[i]] = acc[i][mt][r] + bias;
    }
  }
  __syncthreads();

  // ---- step 3: per-(row,head) softmax over d=64 -> Abf (bf16); 512 segments ----
  for (int it = 0; it < 2; ++it) {
    int seg = tid + 256 * it;
    int m = seg >> 3, h = seg & 7;
    const float* qrow = Fq + m * FQS + h * DH;
    float mx = -3.0e38f;
    for (int i = 0; i < DH; ++i) mx = fmaxf(mx, qrow[i]);
    float s = 0.f;
    for (int i = 0; i < DH; ++i) s += __expf(qrow[i] - mx);
    float invs = 1.0f / s;
    unsigned short* arow = Abf + m * AST + h * DH;
    for (int i = 0; i < DH; ++i) arow[i] = f2bf(__expf(qrow[i] - mx) * invs);
  }
  __syncthreads();

  // ---- step 4: y = qsm @ attT^T per head (WMMA) -> Fq; wave == head ----
  {
    const int h = wav;
    const unsigned short* attB = attT + (size_t)(b * HH + h) * DH * DH;
    for (int pass = 0; pass < 2; ++pass) {
      v8f acc[2][MT] = {};
      #pragma unroll
      for (int kc = 0; kc < 2; ++kc) {
        v16bf a[MT];
        #pragma unroll
        for (int mt = 0; mt < MT; ++mt)
          a[mt] = load_fragA(Abf + (mt * 16 + rsel) * AST + h * DH + kc * 32 + asel);
        #pragma unroll
        for (int i = 0; i < 2; ++i) {
          int nt = pass * 2 + i;
          v16bf bm = load_fragB(attB + (nt * 16 + rsel) * DH + kc * 32 + bsel);
          #pragma unroll
          for (int mt = 0; mt < MT; ++mt)
            acc[i][mt] = wmma_bf16(a[mt], bm, acc[i][mt]);
        }
      }
      #pragma unroll
      for (int i = 0; i < 2; ++i) {
        int n = h * DH + (pass * 2 + i) * 16 + rsel;
        #pragma unroll
        for (int mt = 0; mt < MT; ++mt)
          #pragma unroll
          for (int r = 0; r < 8; ++r)
            Fq[(mt * 16 + mbase + r) * FQS + n] = acc[i][mt][r];
      }
    }
  }
  __syncthreads();

  // ---- step 5: h = silu(LN(y)*(1+scale)+shift) -> Abf (bf16) ----
  {
    const int r = tid >> 2, seg = tid & 3;
    const float* fr = Fq + r * FQS;
    float s = 0.f, ss = 0.f;
    for (int j = 0; j < 128; ++j) { float v = fr[seg + 4 * j]; s += v; ss += v * v; }
    s += __shfl_xor(s, 1, 4); ss += __shfl_xor(ss, 1, 4);
    s += __shfl_xor(s, 2, 4); ss += __shfl_xor(ss, 2, 4);
    float mean = s * (1.0f / DD);
    float var  = ss * (1.0f / DD) - mean * mean;
    float inv  = rsqrtf(var + 1e-5f);
    const float* eob = eo + (size_t)b * 1024;
    unsigned short* ar = Abf + r * AST;
    for (int j = 0; j < 128; ++j) {
      int c = seg + 4 * j;
      float hv = (fr[c] - mean) * inv * sn_g[c] + sn_b[c];
      hv = hv * (1.0f + eob[c]) + eob[DD + c];
      ar[c] = f2bf(hv / (1.0f + __expf(-hv)));
    }
  }
  __syncthreads();

  // ---- step 6: out = x + h @ out_W + out_b (WMMA, ping-pong pipelined) ----
  for (int pass = 0; pass < 2; ++pass) {
    v8f acc[2][MT] = {};
    int ncol[2];
    #pragma unroll
    for (int i = 0; i < 2; ++i) ncol[i] = (wav + 8 * (pass * 2 + i)) * 16 + rsel;
    const unsigned short* arow0 = Abf + rsel * AST + asel;
    const unsigned short* bp[2];
    #pragma unroll
    for (int i = 0; i < 2; ++i) bp[i] = OutWT + (size_t)ncol[i] * DD + bsel;

    v16bf aP[MT], aQ[MT], bP[2], bQ[2];
    #pragma unroll
    for (int mt = 0; mt < MT; ++mt) aP[mt] = load_fragA(arow0 + mt * 16 * AST);
    #pragma unroll
    for (int i = 0; i < 2; ++i) bP[i] = load_fragB(bp[i]);

    for (int kc = 0; kc < DD / 32; kc += 2) {
      const int kb1 = (kc + 1) * 32;
      #pragma unroll
      for (int mt = 0; mt < MT; ++mt) aQ[mt] = load_fragA(arow0 + mt * 16 * AST + kb1);
      #pragma unroll
      for (int i = 0; i < 2; ++i) bQ[i] = load_fragB(bp[i] + kb1);
      #pragma unroll
      for (int i = 0; i < 2; ++i)
        #pragma unroll
        for (int mt = 0; mt < MT; ++mt) acc[i][mt] = wmma_bf16(aP[mt], bP[i], acc[i][mt]);
      if (kc + 2 < DD / 32) {
        const int kb2 = (kc + 2) * 32;
        #pragma unroll
        for (int mt = 0; mt < MT; ++mt) aP[mt] = load_fragA(arow0 + mt * 16 * AST + kb2);
        #pragma unroll
        for (int i = 0; i < 2; ++i) bP[i] = load_fragB(bp[i] + kb2);
      }
      #pragma unroll
      for (int i = 0; i < 2; ++i)
        #pragma unroll
        for (int mt = 0; mt < MT; ++mt) acc[i][mt] = wmma_bf16(aQ[mt], bQ[i], acc[i][mt]);
    }
    #pragma unroll
    for (int i = 0; i < 2; ++i) {
      float bias = out_b[ncol[i]];
      #pragma unroll
      for (int mt = 0; mt < MT; ++mt)
        #pragma unroll
        for (int r = 0; r < 8; ++r) {
          size_t gi = (size_t)(row0 + mt * 16 + mbase + r) * DD + ncol[i];
          out[gi] = x[gi] + acc[i][mt][r] + bias;
        }
    }
  }
}

// ---------------- host launch ----------------
extern "C" void kernel_launch(void* const* d_in, const int* in_sizes, int n_in,
                              void* d_out, int out_size, void* d_ws, size_t ws_size,
                              hipStream_t stream) {
  const float* x     = (const float*)d_in[0];
  const float* xf    = (const float*)d_in[1];
  const float* emb   = (const float*)d_in[2];
  const float* Wq    = (const float*)d_in[3];
  const float* bq    = (const float*)d_in[4];
  const float* Wk    = (const float*)d_in[5];
  const float* bk    = (const float*)d_in[6];
  const float* Wv    = (const float*)d_in[7];
  const float* bv    = (const float*)d_in[8];
  const float* ln_g  = (const float*)d_in[9];
  const float* ln_b  = (const float*)d_in[10];
  const float* tln_g = (const float*)d_in[11];
  const float* tln_b = (const float*)d_in[12];
  const float* embW  = (const float*)d_in[13];
  const float* embB  = (const float*)d_in[14];
  const float* sn_g  = (const float*)d_in[15];
  const float* sn_b  = (const float*)d_in[16];
  const float* outW  = (const float*)d_in[17];
  const float* outb  = (const float*)d_in[18];
  float* out = (float*)d_out;

  char* ws = (char*)d_ws;
  unsigned short* WqT   = (unsigned short*)ws; ws += (size_t)DD * DD * 2;
  unsigned short* OutWT = (unsigned short*)ws; ws += (size_t)DD * DD * 2;
  unsigned short* WkT   = (unsigned short*)ws; ws += (size_t)DD * LL * 2;
  unsigned short* WvT   = (unsigned short*)ws; ws += (size_t)DD * LL * 2;
  unsigned short* attT  = (unsigned short*)ws; ws += (size_t)BB * HH * DH * DH * 2;
  float* eo   = (float*)ws; ws += (size_t)BB * 1024 * 4;
  float* kbuf = (float*)ws; ws += (size_t)BB * NN * DD * 4;
  float* vbuf = (float*)ws; ws += (size_t)BB * NN * DD * 4;
  unsigned short* xfn = (unsigned short*)ws; ws += (size_t)BB * NN * LL * 2;

  k_transpose_bf16<<<(DD * DD + 255) / 256, 256, 0, stream>>>(Wq,   WqT,   DD, DD);
  k_transpose_bf16<<<(DD * DD + 255) / 256, 256, 0, stream>>>(outW, OutWT, DD, DD);
  k_transpose_bf16<<<(LL * DD + 255) / 256, 256, 0, stream>>>(Wk,   WkT,   LL, DD);
  k_transpose_bf16<<<(LL * DD + 255) / 256, 256, 0, stream>>>(Wv,   WvT,   LL, DD);
  k_emb<<<(BB * 1024) / 256, 256, 0, stream>>>(emb, embW, embB, eo);
  k_xfn<<<BB * NN, 256, 0, stream>>>(xf, tln_g, tln_b, xfn);
  k_kv<<<dim3((BB * NN) / 16, 2), 256, 0, stream>>>(xfn, WkT, WvT, bk, bv, kbuf, vbuf);
  k_softk<<<(BB * DD) / 256, 256, 0, stream>>>(kbuf);
  k_att<<<(BB * HH * DH * DH) / 256, 256, 0, stream>>>(kbuf, vbuf, attT);
  fused_main<<<(BB * TT) / ROWS, 256, 0, stream>>>(x, bq, ln_g, ln_b, sn_g, sn_b, outb,
                                                   WqT, OutWT, attT, eo, out);
}
